// ResConvSelfAttnBlock_120259084634
// MI455X (gfx1250) — matrix operand order repaired and checked by reading, not compiled, not run on hardware
//
#include <hip/hip_runtime.h>
#include <hip/hip_bf16.h>
#include <math.h>

// ---------------------------------------------------------------------------
// ResConvSelfAttnBlock for MI455X (gfx1250, wave32, WMMA bf16 16x16x32)
//   x:(16,512,32,32) f32 -> GroupNorm(32) -> qkv 1x1 -> full self-attn over
//   N=1024 positions -> out 1x1 -> +residual
// Matmuls: v_wmma_f32_16x16x32_bf16 (f32 accumulate).
// Weight tiles staged to LDS via TENSOR_LOAD_TO_LDS (TDM) when available.
// ---------------------------------------------------------------------------

typedef __bf16 bf16_t;
typedef __attribute__((ext_vector_type(16))) __bf16 v16bf;
typedef __attribute__((ext_vector_type(8)))  __bf16 v8bf;
typedef __attribute__((ext_vector_type(8)))  float  v8f;
typedef __attribute__((ext_vector_type(4)))  unsigned int u32x4;
typedef __attribute__((ext_vector_type(8)))  int i32x8;
typedef __attribute__((ext_vector_type(4)))  int i32x4;

#define WMMA_BF16(a, b, c) \
  __builtin_amdgcn_wmma_f32_16x16x32_bf16(false, (a), false, (b), (short)0, (c), false, false)

static constexpr int B_   = 16;
static constexpr int C_   = 512;
static constexpr int NSP  = 1024;   // h*w
static constexpr int TC   = 1536;   // 3*C
static constexpr int NGRP = 32;
static constexpr int CPG  = 16;     // C / NGRP

__device__ __forceinline__ v16bf make_v16(v8bf lo, v8bf hi) {
  return __builtin_shufflevector(lo, hi, 0,1,2,3,4,5,6,7,8,9,10,11,12,13,14,15);
}

// A fragment: A[m,k] row-major, rows contiguous in K.
// A 16x32 layout (contiguous k, matching B): lanes 0-15 hold row M=lane, K=0..15;
// lanes 16-31 hold row M=lane-16, K=16..31.
__device__ __forceinline__ v16bf load_a_frag(const bf16_t* base, int lda) {
  int lane = threadIdx.x & 31;
  const bf16_t* p = base + (lane & 15) * lda + (lane >> 4) * 16;
  v8bf lo = *reinterpret_cast<const v8bf*>(p);        // K = half*16 + 0..7
  v8bf hi = *reinterpret_cast<const v8bf*>(p + 8);    // K = half*16 + 8..15
  return make_v16(lo, hi);
}

// B fragment from B^T storage (BT[n][k], K contiguous).
// 16-bit B 32x16 (contiguous k, matching A): lanes 0-15 hold col N=lane, K=0..15; lanes 16-31 K=16..31.
__device__ __forceinline__ v16bf load_bt_frag(const bf16_t* baseT, int ldbt) {
  int lane = threadIdx.x & 31;
  const bf16_t* p = baseT + (lane & 15) * ldbt + (lane >> 4) * 16;
  v8bf lo = *reinterpret_cast<const v8bf*>(p);
  v8bf hi = *reinterpret_cast<const v8bf*>(p + 8);
  return make_v16(lo, hi);
}

// ---------------------------------------------------------------------------
// TDM: 1-D copy of a contiguous span (n8 x 8 bytes) from global to LDS.
// D# per CDNA5 ISA ch.8: group0 = {count/flags, lds_addr, global_addr, type=2},
// group1 = {data_size=8B, tensor_dim0, tensor_dim1=1, tile_dim0}; 1-D tile.
// Issue from ONE wave, that wave waits TENSORcnt, then workgroup barrier.
// ---------------------------------------------------------------------------
#if __has_builtin(__builtin_amdgcn_tensor_load_to_lds)
#define HAVE_TDM 1
__device__ __forceinline__ void tdm_load_1d(unsigned lds_addr, const void* gptr,
                                            unsigned n8 /* 8-byte units */) {
  unsigned long long ga = (unsigned long long)(uintptr_t)gptr;
  u32x4 g0;
  g0[0] = 1u;                                        // count=1, user mode
  g0[1] = lds_addr;                                  // LDS byte address
  g0[2] = (unsigned)(ga & 0xffffffffu);              // global_addr[31:0]
  g0[3] = (unsigned)((ga >> 32) & 0x01ffffffu) | (2u << 30);  // addr[56:32]|type=2
  i32x8 g1;
  g1[0] = (int)(3u << 16);                           // wg_mask=0, data_size=3 (8B)
  g1[1] = (int)((n8 & 0xffffu) << 16);               // tensor_dim0[15:0] @ bits[63:48]
  g1[2] = (int)(((n8 >> 16) & 0xffffu) | (1u << 16));// tensor_dim0[31:16], tensor_dim1=1
  g1[3] = (int)((n8 & 0xffffu) << 16);               // tile_dim0 @ bits[127:112]
  g1[4] = 0; g1[5] = 0; g1[6] = 0; g1[7] = 0;        // tile_dim1/2=0 (1-D), strides=0
  i32x4 z4 = {0, 0, 0, 0};
#if __clang_major__ >= 23
  i32x8 z8 = {0, 0, 0, 0, 0, 0, 0, 0};
  __builtin_amdgcn_tensor_load_to_lds(g0, g1, z4, z4, z8, 0);
#else
  __builtin_amdgcn_tensor_load_to_lds(g0, g1, z4, z4, 0);
#endif
}
#else
#define HAVE_TDM 0
#endif

// Stage a contiguous 64KB (64 rows x 512 bf16) weight tile into LDS.
__device__ __forceinline__ void stage_wtile(bf16_t* lds_tile, const bf16_t* gsrc) {
#if HAVE_TDM
  if ((threadIdx.x >> 5) == 0) {
    tdm_load_1d((unsigned)(uintptr_t)lds_tile, gsrc, (64 * C_ * 2) / 8);
    __builtin_amdgcn_s_wait_tensorcnt(0);
  }
#else
  uint4* d = reinterpret_cast<uint4*>(lds_tile);
  const uint4* s = reinterpret_cast<const uint4*>(gsrc);
  for (int i = threadIdx.x; i < (64 * C_ * 2) / 16; i += 256) d[i] = s[i];
#endif
  __syncthreads();
}

// ---------------------------------------------------------------------------
// K1: GroupNorm -> bf16, stored transposed xnT[b][n][c]
// ---------------------------------------------------------------------------
__global__ __launch_bounds__(256) void gn_kernel(const float* __restrict__ x,
                                                 const float* __restrict__ gamma,
                                                 const float* __restrict__ beta,
                                                 bf16_t* __restrict__ xnT) {
  int b  = blockIdx.x / NGRP;
  int g  = blockIdx.x % NGRP;
  int c0 = g * CPG;
  int t  = threadIdx.x;

  __shared__ float s_sum[256], s_sq[256];
  __shared__ float s_mean, s_rstd;

  float sum = 0.f, sq = 0.f;
  for (int i = 0; i < 64; ++i) {
    int e  = t + i * 256;
    int cl = e >> 10;
    int n  = e & 1023;
    float v = x[((b * C_ + c0 + cl) << 10) + n];
    sum += v; sq += v * v;
  }
  s_sum[t] = sum; s_sq[t] = sq;
  __syncthreads();
  for (int s = 128; s > 0; s >>= 1) {
    if (t < s) { s_sum[t] += s_sum[t + s]; s_sq[t] += s_sq[t + s]; }
    __syncthreads();
  }
  if (t == 0) {
    float m   = s_sum[0] * (1.f / 16384.f);
    float var = s_sq[0] * (1.f / 16384.f) - m * m;
    s_mean = m;
    s_rstd = rsqrtf(var + 1e-5f);
  }
  __syncthreads();
  float mean = s_mean, rstd = s_rstd;

  for (int i = 0; i < 64; ++i) {
    int e  = t + i * 256;
    int cl = e & 15;
    int n  = e >> 4;
    int c  = c0 + cl;
    float v  = x[((b * C_ + c) << 10) + n];
    float xn = (v - mean) * rstd * gamma[c] + beta[c];
    xnT[((b << 10) + n) * C_ + c] = (bf16_t)xn;
  }
}

// ---------------------------------------------------------------------------
// K2: weight conversion fp32 -> bf16
// ---------------------------------------------------------------------------
__global__ __launch_bounds__(256) void cvt_w_kernel(const float* __restrict__ qkvw,
                                                    const float* __restrict__ outw,
                                                    bf16_t* __restrict__ wq,
                                                    bf16_t* __restrict__ wo) {
  int i = blockIdx.x * 256 + threadIdx.x;
  const int n1 = TC * C_;
  const int n2 = C_ * C_;
  if (i < n1)            wq[i]      = (bf16_t)qkvw[i];
  else if (i < n1 + n2)  wo[i - n1] = (bf16_t)outw[i - n1];
}

// ---------------------------------------------------------------------------
// K3: QKV GEMM. block tile 64M x 256N, 8 waves each 32M x 64N.
// A (weights) tile staged to LDS by TDM, shared by all waves.
// q,k stored transposed (n,c); v stored natural (c,n). All bf16.
// ---------------------------------------------------------------------------
__global__ __launch_bounds__(256) void qkv_gemm_kernel(const bf16_t* __restrict__ wq,
                                                       const float* __restrict__ qkvb,
                                                       const bf16_t* __restrict__ xnT,
                                                       bf16_t* __restrict__ qT,
                                                       bf16_t* __restrict__ kT,
                                                       bf16_t* __restrict__ vN) {
  // grid: B * (TC/64) * (NSP/256) = 16 * 24 * 4
  int blk = blockIdx.x;
  int b   = blk / (24 * 4);
  int r   = blk % (24 * 4);
  int m0  = (r / 4) * 64;
  int n0  = (r % 4) * 256;

  __shared__ __align__(16) bf16_t Atile[64 * C_];     // 64KB
  stage_wtile(Atile, wq + (size_t)m0 * C_);

  int wave = threadIdx.x >> 5;
  int wm = wave >> 2;                                 // 0..1 (m offset wm*32)
  int wn = wave & 3;                                  // 0..3 (n offset wn*64)
  const bf16_t* BT = xnT + ((size_t)((b << 10) + n0 + wn * 64)) * C_;

  v8f acc[2][4] = {};
  for (int k = 0; k < C_; k += 32) {
    if (k + 32 < C_) __builtin_prefetch(BT + (k + 32), 0, 0);
    v16bf a0 = load_a_frag(Atile + (wm * 32) * C_ + k, C_);
    v16bf a1 = load_a_frag(Atile + (wm * 32 + 16) * C_ + k, C_);
#pragma unroll
    for (int tt = 0; tt < 4; ++tt) {
      v16bf bb = load_bt_frag(BT + (tt * 16) * C_ + k, C_);
      acc[0][tt] = WMMA_BF16(a0, bb, acc[0][tt]);
      acc[1][tt] = WMMA_BF16(a1, bb, acc[1][tt]);
    }
  }

  int lane = threadIdx.x & 31;
  int nl = lane & 15, mh = (lane >> 4) * 8;
#pragma unroll
  for (int s = 0; s < 2; ++s) {
    int ms = m0 + wm * 32 + s * 16;
#pragma unroll
    for (int tt = 0; tt < 4; ++tt) {
      int n = n0 + wn * 64 + tt * 16 + nl;
#pragma unroll
      for (int rr = 0; rr < 8; ++rr) {
        int m = ms + rr + mh;
        float val = acc[s][tt][rr] + qkvb[m];
        if (m0 < C_)          qT[((b << 10) + n) * C_ + m] = (bf16_t)val;
        else if (m0 < 2 * C_) kT[((b << 10) + n) * C_ + (m - C_)] = (bf16_t)val;
        else                  vN[(b * C_ + (m - 2 * C_)) * NSP + n] = (bf16_t)val;
      }
    }
  }
}

// ---------------------------------------------------------------------------
// K4: fused flash attention.
//   1 workgroup (4 waves, 128 thr) per (batch, 16-row query tile).
//   j in blocks of 128; online softmax in LDS; each wave owns a 128-wide
//   c-slice of the output, accumulated in registers (8 x v8f).
// ---------------------------------------------------------------------------
__global__ __launch_bounds__(128) void flash_attn_kernel(const bf16_t* __restrict__ qT,
                                                         const bf16_t* __restrict__ kT,
                                                         const bf16_t* __restrict__ vN,
                                                         bf16_t* __restrict__ outT) {
  const float SCALE = 0.044194173824159216f;   // 512^-0.5
  const float NEG_BIG = -3.0e38f;

  int b     = blockIdx.x >> 6;
  int it    = blockIdx.x & 63;
  int ibase = it * 16;
  int tid   = threadIdx.x;
  int wave  = tid >> 5;
  int lane  = tid & 31;
  int nl = lane & 15, mh = (lane >> 4) * 8;

  __shared__ __align__(16) float  S[16][128];
  __shared__ __align__(16) bf16_t P[16][128];
  __shared__ __align__(16) bf16_t OST[16][C_];
  __shared__ float red[16][8];
  __shared__ float mrow[16], lrow[16], srow[16];

  if (tid < 16) { mrow[tid] = NEG_BIG; lrow[tid] = 0.f; }

  v8f O[8] = {};
  const bf16_t* qTb = qT + ((size_t)((b << 10) + ibase)) * C_;

  int row = tid & 15;
  int seg = tid >> 4;

  for (int jb = 0; jb < 8; ++jb) {
    int jbase = jb * 128;

    // ---- scores: this wave computes S columns [wave*32, wave*32+32) ----
    v8f sacc[2] = {};
    for (int k = 0; k < C_; k += 32) {
      v16bf a = load_a_frag(qTb + k, C_);
#pragma unroll
      for (int tt = 0; tt < 2; ++tt) {
        const bf16_t* kp = kT + ((size_t)((b << 10) + jbase + wave * 32 + tt * 16)) * C_;
        v16bf bb = load_bt_frag(kp + k, C_);
        sacc[tt] = WMMA_BF16(a, bb, sacc[tt]);
      }
    }
    __syncthreads();
#pragma unroll
    for (int tt = 0; tt < 2; ++tt)
#pragma unroll
      for (int rr = 0; rr < 8; ++rr)
        S[rr + mh][wave * 32 + tt * 16 + nl] = sacc[tt][rr] * SCALE;
    __syncthreads();

    // ---- online softmax: per-row running max ----
    float mx = NEG_BIG;
#pragma unroll
    for (int kk = 0; kk < 16; ++kk) mx = fmaxf(mx, S[row][seg * 16 + kk]);
    red[row][seg] = mx;
    __syncthreads();
    if (tid < 16) {
      float mn = mrow[tid];
#pragma unroll
      for (int s = 0; s < 8; ++s) mn = fmaxf(mn, red[tid][s]);
      srow[tid] = __expf(mrow[tid] - mn);
      mrow[tid] = mn;
    }
    __syncthreads();

    // ---- P = exp(S - m), partial row sums ----
    float mn = mrow[row];
    float ps = 0.f;
#pragma unroll
    for (int kk = 0; kk < 16; ++kk) {
      float p = __expf(S[row][seg * 16 + kk] - mn);
      P[row][seg * 16 + kk] = (bf16_t)p;
      ps += p;
    }
    red[row][seg] = ps;
    __syncthreads();
    if (tid < 16) {
      float l = lrow[tid] * srow[tid];
#pragma unroll
      for (int s = 0; s < 8; ++s) l += red[tid][s];
      lrow[tid] = l;
    }

    // ---- rescale accumulators ----
#pragma unroll
    for (int tt = 0; tt < 8; ++tt)
#pragma unroll
      for (int rr = 0; rr < 8; ++rr)
        O[tt][rr] *= srow[rr + mh];

    // ---- O += P @ V-block ----
#pragma unroll
    for (int ks = 0; ks < 4; ++ks) {
      const bf16_t* pp = &P[lane & 15][ks * 32 + (lane >> 4) * 16];
      v8bf plo = *reinterpret_cast<const v8bf*>(pp);
      v8bf phi = *reinterpret_cast<const v8bf*>(pp + 8);
      v16bf a = make_v16(plo, phi);
#pragma unroll
      for (int tt = 0; tt < 8; ++tt) {
        int c = wave * 128 + tt * 16 + nl;
        const bf16_t* vp = vN + (size_t)(b * C_ + c) * NSP + jbase + ks * 32 + (lane >> 4) * 16;
        v8bf vlo = *reinterpret_cast<const v8bf*>(vp);
        v8bf vhi = *reinterpret_cast<const v8bf*>(vp + 8);
        O[tt] = WMMA_BF16(a, make_v16(vlo, vhi), O[tt]);
      }
    }
    __syncthreads();
  }

  // ---- normalize and write outT[b][i][c] via LDS transpose ----
#pragma unroll
  for (int tt = 0; tt < 8; ++tt)
#pragma unroll
    for (int rr = 0; rr < 8; ++rr) {
      float inv = 1.f / lrow[rr + mh];
      OST[rr + mh][wave * 128 + tt * 16 + nl] = (bf16_t)(O[tt][rr] * inv);
    }
  __syncthreads();
  uint4* dst = reinterpret_cast<uint4*>(outT + (size_t)((b << 10) + ibase) * C_);
  const uint4* src = reinterpret_cast<const uint4*>(&OST[0][0]);
  for (int i = tid; i < (16 * C_ * 2) / 16; i += 128) dst[i] = src[i];
}

// ---------------------------------------------------------------------------
// K5: output projection + bias + residual. Block tile 64M x 256N, TDM-staged A.
// ---------------------------------------------------------------------------
__global__ __launch_bounds__(256) void proj_kernel(const bf16_t* __restrict__ wo,
                                                   const float* __restrict__ outb,
                                                   const bf16_t* __restrict__ outT,
                                                   const float* __restrict__ x,
                                                   float* __restrict__ y) {
  // grid: B * (C/64) * (NSP/256) = 16 * 8 * 4
  int blk = blockIdx.x;
  int b   = blk / (8 * 4);
  int r   = blk % (8 * 4);
  int m0  = (r / 4) * 64;
  int n0  = (r % 4) * 256;

  __shared__ __align__(16) bf16_t Atile[64 * C_];     // 64KB
  stage_wtile(Atile, wo + (size_t)m0 * C_);

  int wave = threadIdx.x >> 5;
  int wm = wave >> 2;
  int wn = wave & 3;
  const bf16_t* BT = outT + ((size_t)((b << 10) + n0 + wn * 64)) * C_;

  v8f acc[2][4] = {};
  for (int k = 0; k < C_; k += 32) {
    if (k + 32 < C_) __builtin_prefetch(BT + (k + 32), 0, 0);
    v16bf a0 = load_a_frag(Atile + (wm * 32) * C_ + k, C_);
    v16bf a1 = load_a_frag(Atile + (wm * 32 + 16) * C_ + k, C_);
#pragma unroll
    for (int tt = 0; tt < 4; ++tt) {
      v16bf bb = load_bt_frag(BT + (tt * 16) * C_ + k, C_);
      acc[0][tt] = WMMA_BF16(a0, bb, acc[0][tt]);
      acc[1][tt] = WMMA_BF16(a1, bb, acc[1][tt]);
    }
  }

  int lane = threadIdx.x & 31;
  int nl = lane & 15, mh = (lane >> 4) * 8;
#pragma unroll
  for (int s = 0; s < 2; ++s) {
    int ms = m0 + wm * 32 + s * 16;
#pragma unroll
    for (int tt = 0; tt < 4; ++tt) {
      int n = n0 + wn * 64 + tt * 16 + nl;
#pragma unroll
      for (int rr = 0; rr < 8; ++rr) {
        int m = ms + rr + mh;
        int idx = (b * C_ + m) * NSP + n;
        y[idx] = acc[s][tt][rr] + outb[m] + x[idx];
      }
    }
  }
}

// ---------------------------------------------------------------------------
extern "C" void kernel_launch(void* const* d_in, const int* in_sizes, int n_in,
                              void* d_out, int out_size, void* d_ws, size_t ws_size,
                              hipStream_t stream) {
  const float* x     = (const float*)d_in[0];
  const float* gnw   = (const float*)d_in[1];
  const float* gnb   = (const float*)d_in[2];
  const float* qkvw  = (const float*)d_in[3];
  const float* qkvb  = (const float*)d_in[4];
  const float* outw  = (const float*)d_in[5];
  const float* outb  = (const float*)d_in[6];
  float* y = (float*)d_out;

  char* base = (char*)d_ws;
  size_t off = 0;
  auto carve = [&](size_t bytes) {
    void* p = base + off;
    off = (off + bytes + 255) & ~(size_t)255;
    return p;
  };
  bf16_t* xnT  = (bf16_t*)carve((size_t)B_ * NSP * C_ * 2);   // 16 MB
  bf16_t* wq   = (bf16_t*)carve((size_t)TC * C_ * 2);
  bf16_t* wo   = (bf16_t*)carve((size_t)C_ * C_ * 2);
  bf16_t* qT   = (bf16_t*)carve((size_t)B_ * NSP * C_ * 2);
  bf16_t* kT   = (bf16_t*)carve((size_t)B_ * NSP * C_ * 2);
  bf16_t* vN   = (bf16_t*)carve((size_t)B_ * C_ * NSP * 2);
  bf16_t* outT = (bf16_t*)carve((size_t)B_ * NSP * C_ * 2);
  (void)ws_size; (void)n_in; (void)in_sizes; (void)out_size;

  gn_kernel<<<B_ * NGRP, 256, 0, stream>>>(x, gnw, gnb, xnT);
  cvt_w_kernel<<<(TC * C_ + C_ * C_) / 256, 256, 0, stream>>>(qkvw, outw, wq, wo);
  qkv_gemm_kernel<<<B_ * (TC / 64) * (NSP / 256), 256, 0, stream>>>(wq, qkvb, xnT, qT, kT, vN);
  flash_attn_kernel<<<B_ * (NSP / 16), 128, 0, stream>>>(qT, kT, vN, outT);
  proj_kernel<<<B_ * (C_ / 64) * (NSP / 256), 256, 0, stream>>>(wo, outb, outT, x, y);
}
